// AttentionModel_61272003444755
// MI455X (gfx1250) — matrix operand order repaired and checked by reading, not compiled, not run on hardware
//
#include <hip/hip_runtime.h>
#include <cmath>

// ---------------------------------------------------------------------------
// Types for CDNA5 WMMA (wave32): D(16x16 f32) = A(16x32 bf16) * B(32x16 bf16) + C
// ---------------------------------------------------------------------------
typedef __attribute__((ext_vector_type(16))) __bf16          v16bf;
typedef __attribute__((ext_vector_type(16))) unsigned short  v16us;
typedef __attribute__((ext_vector_type(8)))  float           v8f;

union BF16V { v16us u; v16bf b; };

__device__ __forceinline__ v8f wmma_bf16(const v16us& a, const v16us& b, v8f c) {
    BF16V A, B;
    A.u = a; B.u = b;
    // 8 args: (neg_a, A, neg_b, B, c_mod, C, reuse_a, reuse_b)
    return __builtin_amdgcn_wmma_f32_16x16x32_bf16(false, A.b, false, B.b,
                                                   (short)0, c, false, false);
}

__device__ __forceinline__ unsigned short f2bf(float f) {
    unsigned int u = __float_as_uint(f);
    u += 0x7FFFu + ((u >> 16) & 1u);   // round-to-nearest-even
    return (unsigned short)(u >> 16);
}

__device__ __forceinline__ float gelu_exact(float x) {
    return 0.5f * x * (1.0f + erff(x * 0.70710678118654752f));
}

// Model constants
#define BB   4
#define NN   64
#define TT   192
#define HH   256
#define TOK  (BB * NN * TT)      // 49152 tokens
#define LMAX 192

// GEMM tiling
#define BM 128
#define BN 128
#define BK 32

// ---------------------------------------------------------------------------
// start: out[t,h] = x[t,0]*w[h,0] + x[t,1]*w[h,1] + b[h]   (K=2, VALU is right)
// ---------------------------------------------------------------------------
__global__ void start_kernel(const float* __restrict__ x, const float* __restrict__ w,
                             const float* __restrict__ b, float* __restrict__ out) {
    int i = blockIdx.x * blockDim.x + threadIdx.x;
    if (i >= TOK * HH) return;
    int tok = i >> 8, h = i & 255;
    out[i] = x[tok * 2] * w[h * 2] + x[tok * 2 + 1] * w[h * 2 + 1] + b[h];
}

// ---------------------------------------------------------------------------
// WMMA GEMM: C[M,N] = act(A[M,K] @ W[N,K]^T + bias[N]) (+ resid[M,N])
// block = 256 threads (8 waves), 128x128 C tile; each wave owns 32x64
// (2x4 WMMA tiles = 8 wmma per K-chunk). K staged in 32-chunks, double-
// buffered LDS so global loads of chunk k+1 overlap WMMA of chunk k.
// M,N,K multiples of 128/32 in this model.
// ---------------------------------------------------------------------------
__global__ __launch_bounds__(256) void
gemm_kernel(const float* __restrict__ A, const float* __restrict__ W,
            const float* __restrict__ bias, const float* __restrict__ resid,
            float* __restrict__ C, int M, int N, int K, int act) {
    __shared__ unsigned short As[2][BM * BK];
    __shared__ unsigned short Bs[2][BN * BK];

    const int tid  = threadIdx.x;
    const int wave = tid >> 5, lane = tid & 31;
    const int wr = wave >> 1;                 // 0..3 : 32-row block
    const int wc = wave & 1;                  // 0..1 : 64-col block
    const int g = lane >> 4, q = lane & 15;   // lane half / lane index
    const int m0 = blockIdx.x * BM, n0 = blockIdx.y * BN;

    v8f acc[2][4];
#pragma unroll
    for (int i = 0; i < 2; ++i)
#pragma unroll
        for (int j = 0; j < 4; ++j) acc[i][j] = 0.0f;

    const int srow = tid >> 1;                // 0..127
    const int skh  = (tid & 1) * 16;

    auto stage = [&](int buf, int k0) {
        const float* sa = A + (size_t)(m0 + srow) * K + k0 + skh;
        const float* sw = W + (size_t)(n0 + srow) * K + k0 + skh;
        // speculative prefetch of the chunk after next (global_prefetch_b8)
        __builtin_prefetch((const void*)(sa + BK), 0, 0);
        __builtin_prefetch((const void*)(sw + BK), 0, 0);
#pragma unroll
        for (int i = 0; i < 16; i += 4) {
            float4 fa = *(const float4*)(sa + i);
            float4 fw = *(const float4*)(sw + i);
            As[buf][srow * BK + skh + i + 0] = f2bf(fa.x);
            As[buf][srow * BK + skh + i + 1] = f2bf(fa.y);
            As[buf][srow * BK + skh + i + 2] = f2bf(fa.z);
            As[buf][srow * BK + skh + i + 3] = f2bf(fa.w);
            Bs[buf][srow * BK + skh + i + 0] = f2bf(fw.x);
            Bs[buf][srow * BK + skh + i + 1] = f2bf(fw.y);
            Bs[buf][srow * BK + skh + i + 2] = f2bf(fw.z);
            Bs[buf][srow * BK + skh + i + 3] = f2bf(fw.w);
        }
    };

    stage(0, 0);
    for (int k0 = 0; k0 < K; k0 += BK) {
        const int cur = (k0 >> 5) & 1;
        __syncthreads();                      // staging of `cur` complete

        v16us afrag[2], bfrag[4];
#pragma unroll
        for (int mt = 0; mt < 2; ++mt) {
            int rowA = wr * 32 + mt * 16 + q;
#pragma unroll
            for (int v = 0; v < 8; ++v) {
                // A-matrix: lane half g covers 16 contiguous K values (matches B)
                int kb = g * 16 + 2 * v;
                afrag[mt][2 * v]     = As[cur][rowA * BK + kb];
                afrag[mt][2 * v + 1] = As[cur][rowA * BK + kb + 1];
            }
        }
#pragma unroll
        for (int nt = 0; nt < 4; ++nt) {
            int rowB = wc * 64 + nt * 16 + q;
#pragma unroll
            for (int v = 0; v < 8; ++v) {
                // B-matrix: lane half g covers 16 contiguous K values
                int kb = g * 16 + 2 * v;
                bfrag[nt][2 * v]     = Bs[cur][rowB * BK + kb];
                bfrag[nt][2 * v + 1] = Bs[cur][rowB * BK + kb + 1];
            }
        }

        if (k0 + BK < K) stage(cur ^ 1, k0 + BK);   // overlap with WMMA below

#pragma unroll
        for (int mt = 0; mt < 2; ++mt)
#pragma unroll
            for (int nt = 0; nt < 4; ++nt)
                acc[mt][nt] = wmma_bf16(afrag[mt], bfrag[nt], acc[mt][nt]);
    }

#pragma unroll
    for (int mt = 0; mt < 2; ++mt)
#pragma unroll
        for (int nt = 0; nt < 4; ++nt)
#pragma unroll
            for (int r = 0; r < 8; ++r) {
                int m = m0 + wr * 32 + mt * 16 + g * 8 + r;
                int n = n0 + wc * 64 + nt * 16 + q;
                float v = acc[mt][nt][r] + bias[n];
                if (act) v = gelu_exact(v);
                size_t idx = (size_t)m * N + n;
                if (resid) v += resid[idx];
                C[idx] = v;
            }
}

// ---------------------------------------------------------------------------
// Fused attention for one (sequence, 16-query block).
// mode 0 (temporal): seq s in [0, B*N), token(l) = s*T + l          (stride 1)
// mode 1 (spatial) : seq s in [0, B*T), token(l) = (s/T)*N*T + s%T + l*T
// L = 192 (mode 0) or 64 (mode 1); H = 256. block = 128 threads (4 waves).
// ---------------------------------------------------------------------------
__global__ __launch_bounds__(128) void
attn_kernel(const float* __restrict__ Q, const float* __restrict__ K,
            const float* __restrict__ V, float* __restrict__ O,
            int L, int mode) {
    __shared__ unsigned short Qs[16 * HH];     //  8 KB, query block bf16
    __shared__ float          Sc[16 * LMAX];   // 12 KB, f32 scores
    __shared__ unsigned short Ps[16 * LMAX];   //  6 KB, softmax probs bf16
    __shared__ unsigned short Vt[HH * 32];     // 16 KB, transposed V chunk [h][k]

    const int s  = blockIdx.x;
    const int qb = blockIdx.y;
    long base; int stride;
    if (mode == 0) { base = (long)s * TT;                        stride = 1;  }
    else           { base = (long)(s / TT) * NN * TT + (s % TT); stride = TT; }

    const int tid  = threadIdx.x;
    const int wave = tid >> 5, lane = tid & 31;
    const int g = lane >> 4, qn = lane & 15;

    // stage Q rows (16 x 256) -> LDS bf16
    for (int e = tid; e < 16 * HH; e += 128) {
        int row = e >> 8, k = e & 255;
        long tok = base + (long)(qb * 16 + row) * stride;
        Qs[row * HH + k] = f2bf(Q[tok * HH + k]);
    }
    __syncthreads();

    // ---- scores: S[16, L] = (Qblk @ K^T) / 16, key tiles striped over waves
    const int nkt = L >> 4;
    for (int j = wave; j < nkt; j += 4) {
        v8f acc = 0.0f;
        long ktok = base + (long)(j * 16 + qn) * stride;   // this lane's key row
        const float* krow = K + ktok * HH;
        for (int kk = 0; kk < HH; kk += 32) {
            v16us au, bu;
#pragma unroll
            for (int v = 0; v < 8; ++v) {
                int ka = g * 16 + 2 * v;
                au[2 * v]     = Qs[qn * HH + kk + ka];
                au[2 * v + 1] = Qs[qn * HH + kk + ka + 1];
                int kb = g * 16 + 2 * v;
                float2 kv = *(const float2*)(krow + kk + kb);
                bu[2 * v]     = f2bf(kv.x);
                bu[2 * v + 1] = f2bf(kv.y);
            }
            acc = wmma_bf16(au, bu, acc);
        }
#pragma unroll
        for (int r = 0; r < 8; ++r) {
            int m = g * 8 + r;
            Sc[m * LMAX + j * 16 + qn] = acc[r] * 0.0625f;   // / sqrt(256)
        }
    }
    __syncthreads();

    // ---- softmax over L: 8 lanes per row (rows are contiguous 8-lane groups)
    {
        int row = tid >> 3, sub = tid & 7;
        float mx = -1e30f;
        for (int n = sub; n < L; n += 8) mx = fmaxf(mx, Sc[row * LMAX + n]);
        mx = fmaxf(mx, __shfl_xor(mx, 1, 32));
        mx = fmaxf(mx, __shfl_xor(mx, 2, 32));
        mx = fmaxf(mx, __shfl_xor(mx, 4, 32));
        float sum = 0.0f;
        for (int n = sub; n < L; n += 8) sum += expf(Sc[row * LMAX + n] - mx);
        sum += __shfl_xor(sum, 1, 32);
        sum += __shfl_xor(sum, 2, 32);
        sum += __shfl_xor(sum, 4, 32);
        float inv = 1.0f / sum;
        for (int n = sub; n < L; n += 8)
            Ps[row * LMAX + n] = f2bf(expf(Sc[row * LMAX + n] - mx) * inv);
    }
    __syncthreads();

    // ---- O[16, 256] = P[16, L] @ V[L, 256]
    // K-chunks of 32: V chunk staged transposed into LDS (coalesced global
    // reads, contiguous ds_load_b128 fragment fetch); wave owns 64 columns.
    v8f oacc[4];
#pragma unroll
    for (int nt = 0; nt < 4; ++nt) oacc[nt] = 0.0f;
    const int h0w = wave * 64;

    for (int kk = 0; kk < L; kk += 32) {
        for (int e = tid; e < 32 * HH; e += 128) {
            int kr = e >> 8, h = e & 255;
            long tok = base + (long)(kk + kr) * stride;
            Vt[h * 32 + kr] = f2bf(V[tok * HH + h]);
        }
        __syncthreads();

        v16us au;
#pragma unroll
        for (int v = 0; v < 8; ++v) {
            int ka = g * 16 + 2 * v;
            au[2 * v]     = Ps[qn * LMAX + kk + ka];
            au[2 * v + 1] = Ps[qn * LMAX + kk + ka + 1];
        }
#pragma unroll
        for (int nt = 0; nt < 4; ++nt) {
            v16us bu;
            int h = h0w + nt * 16 + qn;
#pragma unroll
            for (int v = 0; v < 8; ++v) {
                int kb = g * 16 + 2 * v;
                bu[2 * v]     = Vt[h * 32 + kb];
                bu[2 * v + 1] = Vt[h * 32 + kb + 1];
            }
            oacc[nt] = wmma_bf16(au, bu, oacc[nt]);
        }
        __syncthreads();
    }

#pragma unroll
    for (int nt = 0; nt < 4; ++nt)
#pragma unroll
        for (int r = 0; r < 8; ++r) {
            int m = g * 8 + r;
            long tok = base + (long)(qb * 16 + m) * stride;
            O[tok * HH + h0w + nt * 16 + qn] = oacc[nt][r];
        }
}

// ---------------------------------------------------------------------------
// LayerNorm rows of H=256: (x - mean) / (std + eps) * g + b, one wave per row
// ---------------------------------------------------------------------------
__global__ __launch_bounds__(256) void
ln_kernel(const float* __restrict__ X, const float* __restrict__ gam,
          const float* __restrict__ bet, float* __restrict__ Y) {
    int wave = threadIdx.x >> 5, lane = threadIdx.x & 31;
    int row = blockIdx.x * 8 + wave;
    const float* xr = X + (size_t)row * HH;
    float v[8];
    float s = 0.0f;
#pragma unroll
    for (int i = 0; i < 8; ++i) { v[i] = xr[lane + i * 32]; s += v[i]; }
#pragma unroll
    for (int off = 16; off > 0; off >>= 1) s += __shfl_xor(s, off, 32);
    float mean = s * (1.0f / 256.0f);
    float ss = 0.0f;
#pragma unroll
    for (int i = 0; i < 8; ++i) { float d = v[i] - mean; ss += d * d; }
#pragma unroll
    for (int off = 16; off > 0; off >>= 1) ss += __shfl_xor(ss, off, 32);
    float inv = 1.0f / (sqrtf(ss * (1.0f / 256.0f)) + 1e-6f);
    float* yr = Y + (size_t)row * HH;
#pragma unroll
    for (int i = 0; i < 8; ++i) {
        int k = lane + i * 32;
        yr[k] = (v[i] - mean) * inv * gam[k] + bet[k];
    }
}

// ---------------------------------------------------------------------------
// final: out[0:total) = relu(X), out[total:2*total) = X
// ---------------------------------------------------------------------------
__global__ void final_kernel(const float* __restrict__ X, float* __restrict__ out,
                             long total) {
    long i = (long)blockIdx.x * blockDim.x + threadIdx.x;
    if (i >= total) return;
    float v = X[i];
    out[i] = fmaxf(v, 0.0f);
    out[total + i] = v;
}

// ---------------------------------------------------------------------------
// Host orchestration.
// d_in order (insertion order of setup_inputs): x, start.w, start.b, then per
// layer: t_attn{q,k,v,p}{w,b}(8), ln_t{g,b}(2), s_attn{..}(8), ln_s(2),
// ff1{w,b}(2), ff2{w,b}(2), ln_f{g,b}(2)  => 26 per layer.
// ---------------------------------------------------------------------------
extern "C" void kernel_launch(void* const* d_in, const int* in_sizes, int n_in,
                              void* d_out, int out_size, void* d_ws, size_t ws_size,
                              hipStream_t stream) {
    (void)in_sizes; (void)n_in; (void)out_size; (void)ws_size;

    float* ws = (float*)d_ws;
    float* X  = ws;                                // [TOK, 256]
    float* Y  = X + (size_t)TOK * HH;              // [TOK, 256]
    float* Fb = Y + (size_t)TOK * HH;              // [TOK, 1024] == Q|K|V|Ctx
    float* Qb = Fb;
    float* Kb = Fb + (size_t)TOK * HH;
    float* Vb = Fb + 2 * (size_t)TOK * HH;
    float* Cb = Fb + 3 * (size_t)TOK * HH;

    start_kernel<<<(TOK * HH) / 256, 256, 0, stream>>>(
        (const float*)d_in[0], (const float*)d_in[1], (const float*)d_in[2], X);

    const dim3 g256(TOK / BM, HH / BN);            // (384, 2)
    const dim3 g1024(TOK / BM, 1024 / BN);         // (384, 8)

    for (int l = 0; l < 3; ++l) {
        int p = 3 + l * 26;
        auto P = [&](int i) { return (const float*)d_in[p + i]; };

        // ---- temporal attention + LN
        gemm_kernel<<<g256, 256, 0, stream>>>(X, P(0), P(1), nullptr, Qb, TOK, HH, HH, 0);
        gemm_kernel<<<g256, 256, 0, stream>>>(X, P(2), P(3), nullptr, Kb, TOK, HH, HH, 0);
        gemm_kernel<<<g256, 256, 0, stream>>>(X, P(4), P(5), nullptr, Vb, TOK, HH, HH, 0);
        attn_kernel<<<dim3(BB * NN, TT / 16), 128, 0, stream>>>(Qb, Kb, Vb, Cb, TT, 0);
        gemm_kernel<<<g256, 256, 0, stream>>>(Cb, P(6), P(7), X, Qb, TOK, HH, HH, 0);
        ln_kernel<<<TOK / 8, 256, 0, stream>>>(Qb, P(8), P(9), Y);

        // ---- spatial attention + LN (logical transpose via strided indexing)
        gemm_kernel<<<g256, 256, 0, stream>>>(Y, P(10), P(11), nullptr, Qb, TOK, HH, HH, 0);
        gemm_kernel<<<g256, 256, 0, stream>>>(Y, P(12), P(13), nullptr, Kb, TOK, HH, HH, 0);
        gemm_kernel<<<g256, 256, 0, stream>>>(Y, P(14), P(15), nullptr, Vb, TOK, HH, HH, 0);
        attn_kernel<<<dim3(BB * TT, NN / 16), 128, 0, stream>>>(Qb, Kb, Vb, Cb, NN, 1);
        gemm_kernel<<<g256, 256, 0, stream>>>(Cb, P(16), P(17), Y, Qb, TOK, HH, HH, 0);
        ln_kernel<<<TOK / 8, 256, 0, stream>>>(Qb, P(18), P(19), X);

        // ---- FFN + LN (Fb aliases the dead Q/K/V/Ctx region)
        gemm_kernel<<<g1024, 256, 0, stream>>>(X, P(20), P(21), nullptr, Fb, TOK, 1024, HH, 1);
        gemm_kernel<<<g256, 256, 0, stream>>>(Fb, P(22), P(23), X, Y, TOK, HH, 1024, 0);
        ln_kernel<<<TOK / 8, 256, 0, stream>>>(Y, P(24), P(25), X);
    }

    final_kernel<<<(TOK * HH) / 256, 256, 0, stream>>>(X, (float*)d_out, (long)TOK * HH);
}
